// BoTBlock_64458869178601
// MI455X (gfx1250) — matrix operand order repaired and checked by reading, not compiled, not run on hardware
//
#include <hip/hip_runtime.h>
#include <hip/hip_bf16.h>

// ---------------------------------------------------------------------------
// Types / WMMA helpers (gfx1250, wave32)
// ---------------------------------------------------------------------------
typedef __attribute__((ext_vector_type(16))) __bf16 v16bf;
typedef __attribute__((ext_vector_type(8)))  float  v8f;
typedef __attribute__((ext_vector_type(4)))  unsigned int v4u;
typedef __attribute__((ext_vector_type(4)))  int v4i;
typedef __attribute__((ext_vector_type(8)))  int v8i32;

union AFrag { v16bf v; unsigned int d[8]; unsigned short h[16]; };

__device__ __forceinline__ unsigned short f2bf(float f) {
  unsigned int u = __builtin_bit_cast(unsigned int, f);
  u += 0x7FFFu + ((u >> 16) & 1u);           // round-to-nearest-even
  return (unsigned short)(u >> 16);
}

__device__ __forceinline__ v8f wmma_bf16(v16bf a, v16bf b, v8f c) {
  return __builtin_amdgcn_wmma_f32_16x16x32_bf16(false, a, false, b, (short)0, c,
                                                 false, false);
}

__device__ __forceinline__ v8f v8zero() {
  v8f r;
#pragma unroll
  for (int i = 0; i < 8; ++i) r[i] = 0.0f;
  return r;
}

// ---------------------------------------------------------------------------
// Tensor Data Mover: 2D tile (2-byte elements) global -> LDS.
// D# per CDNA5 ISA. 1D copies: tile1 = ten1 = 1.
// Issued wave-uniform (guarded to wave 0); in-order per wave, so with
// ping-pong buffers we wait for TENSORcnt <= (#ops just issued).
// ---------------------------------------------------------------------------
__device__ __forceinline__ void tdm_load(unsigned ldsOff, const void* gptr,
                                         unsigned tile0, unsigned tile1,
                                         unsigned ten0, unsigned ten1,
                                         unsigned stride0) {
  unsigned long long ga = (unsigned long long)(size_t)gptr;
  v4u g0;
  v8i32 g1;
  v4i gz4 = {0, 0, 0, 0};
  v8i32 gz8 = {0, 0, 0, 0, 0, 0, 0, 0};
  g0[0] = 1u;                                            // count=1, user mode
  g0[1] = ldsOff;                                        // lds_addr (bytes)
  g0[2] = (unsigned)ga;                                  // global_addr[31:0]
  g0[3] = (unsigned)((ga >> 32) & 0x01FFFFFFull) | 0x80000000u;  // [56:32] | type=2
  g1[0] = 0x00010000;                                    // data_size=1 (2 bytes)
  g1[1] = (int)((ten0 & 0xFFFFu) << 16);                 // tensor_dim0[15:0]
  g1[2] = (int)((ten0 >> 16) | ((ten1 & 0xFFFFu) << 16));// dim0[31:16] | dim1[15:0]
  g1[3] = (int)((ten1 >> 16) | (tile0 << 16));           // dim1[31:16] | tile_dim0
  g1[4] = (int)tile1;                                    // tile_dim1 (tile_dim2=0)
  g1[5] = (int)stride0;                                  // tensor_dim0_stride[31:0]
  g1[6] = 0;
  g1[7] = 0;
  __builtin_amdgcn_tensor_load_to_lds(g0, g1, gz4, gz4, gz8, 0);
}

// A fragment 16x32 from LDS tile [rows][32] (K-contiguous rows).
__device__ __forceinline__ v16bf load_a_frag(const unsigned short* As, int mBase, int lane) {
  AFrag f;
  const unsigned int* row = (const unsigned int*)(As + (size_t)(mBase + (lane & 15)) * 32);
  const int kb = (lane >> 4) * 8;
#pragma unroll
  for (int v = 0; v < 8; ++v) {
    const int k = ((v < 4) ? (2 * v) : (16 + 2 * (v - 4))) + kb;  // even
    f.d[v] = row[k >> 1];
  }
  return f.v;
}

// B fragment 32x16 from LDS tile [cols][32] (K-contiguous per column).
__device__ __forceinline__ v16bf load_b_frag(const unsigned short* Bs, int nBase, int lane) {
  AFrag f;
  const unsigned int* row = (const unsigned int*)(Bs + (size_t)(nBase + (lane & 15)) * 32);
  const int kd = (lane >> 4) * 4;
#pragma unroll
  for (int v = 0; v < 8; ++v) f.d[v] = row[kd + v + ((v < 4) ? 0 : 4)];
  return f.v;
}

// ---------------------------------------------------------------------------
// Problem constants
// ---------------------------------------------------------------------------
#define NIMG  16
#define CIN   2048
#define MID   512
#define HW    1024
#define NROWS (NIMG * HW)
#define QKV   1536
#define SCALE 0.08838834764831845f

// ---------------------------------------------------------------------------
// Weight convert: w[O][K] f32 -> blocked bf16 [(k/32)*ld + colOff + o][k%32]
// so each 64x32 GEMM B-tile is one contiguous 4KB block (TDM-friendly).
// ---------------------------------------------------------------------------
__global__ void wcvt_kernel(const float* __restrict__ src, unsigned short* __restrict__ dst,
                            int O, int K, int ld, int colOff) {
  int idx = blockIdx.x * 256 + threadIdx.x;
  if (idx >= O * K) return;
  int o = idx / K, k = idx - o * K;
  dst[((size_t)(k >> 5) * ld + colOff + o) * 32 + (k & 31)] = f2bf(src[idx]);
}

// ---------------------------------------------------------------------------
// conv1: out1[row][512] = relu(bn1(x @ w1^T)); x NCHW f32 (channel-strided A)
// grid (256, 8). B tiles via TDM, double-buffered; A staged manually (f32
// convert) each iteration.
// ---------------------------------------------------------------------------
__global__ __launch_bounds__(256) void conv1_gemm(
    const float* __restrict__ x, const unsigned short* __restrict__ w1B,
    const float* __restrict__ g, const float* __restrict__ bta,
    const float* __restrict__ mn, const float* __restrict__ vr,
    unsigned short* __restrict__ out1) {
  __shared__ __align__(16) unsigned short As[64 * 32];
  __shared__ __align__(16) unsigned short Bs[2][64 * 32];
  const int tid = threadIdx.x, lane = tid & 31, wave = tid >> 5;
  const int wm = wave & 3, wn = wave >> 2;
  const int r0 = blockIdx.x * 64, c0 = blockIdx.y * 64;
  const int nI = r0 >> 10, hw0 = r0 & 1023;

  const float* xa = x + (size_t)nI * CIN * HW + hw0;           // +32*HW per iter
  const unsigned short* wt = w1B + (size_t)c0 * 32;            // +MID*32 per iter
  const unsigned bsOff = (unsigned)(size_t)(void*)&Bs[0][0];

  if (wave == 0) tdm_load(bsOff, wt, 2048, 1, 2048, 1, 2048);  // prologue: tile 0

  v8f acc0 = v8zero(), acc1 = v8zero();
  for (int k0 = 0; k0 < CIN; k0 += 32) {
    const int buf = (k0 >> 5) & 1;
#pragma unroll
    for (int i = 0; i < 4; ++i) {      // A: f32 -> bf16, pack pairs along k
      int idx = tid + i * 256;         // 0..1023
      int m = idx & 63, kp2 = idx >> 6;
      float f0 = xa[(size_t)(2 * kp2) * HW + m];
      float f1 = xa[(size_t)(2 * kp2 + 1) * HW + m];
      ((unsigned int*)As)[m * 16 + kp2] =
          (unsigned)f2bf(f0) | ((unsigned)f2bf(f1) << 16);
    }
    if (wave == 0) {
      if (k0 + 32 < CIN) {             // issue next B tile into other buffer
        tdm_load(bsOff + (unsigned)(buf ^ 1) * 4096, wt + (size_t)MID * 32,
                 2048, 1, 2048, 1, 2048);
        __builtin_amdgcn_s_wait_tensorcnt(1);   // current tile complete
      } else {
        __builtin_amdgcn_s_wait_tensorcnt(0);
      }
    }
    __syncthreads();
    v16bf a = load_a_frag(As, wm * 16, lane);
    acc0 = wmma_bf16(a, load_b_frag(Bs[buf], wn * 32, lane), acc0);
    acc1 = wmma_bf16(a, load_b_frag(Bs[buf], wn * 32 + 16, lane), acc1);
    __syncthreads();
    xa += (size_t)32 * HW;
    wt += (size_t)MID * 32;
  }
  const int mrow = r0 + wm * 16 + 8 * (lane >> 4);
#pragma unroll
  for (int f = 0; f < 2; ++f) {
    const int col = c0 + wn * 32 + f * 16 + (lane & 15);
    const float inv = g[col] * rsqrtf(vr[col] + 1e-5f);
    const float add = bta[col] - mn[col] * inv;
    v8f acc = f ? acc1 : acc0;
#pragma unroll
    for (int j = 0; j < 8; ++j) {
      float val = acc[j] * inv + add;
      out1[(size_t)(mrow + j) * MID + col] = f2bf(val > 0.f ? val : 0.f);
    }
  }
}

// ---------------------------------------------------------------------------
// fused q/k/v projection: [16384,512] @ wcat^T; A+B tiles via double-buffered
// TDM. epilogue: q *= SCALE; k += pos_h+pos_w (emb fold); route to qs/kp/vs.
// grid (256, 24)
// ---------------------------------------------------------------------------
__global__ __launch_bounds__(256) void qkv_gemm(
    const unsigned short* __restrict__ out1, const unsigned short* __restrict__ wcB,
    const float* __restrict__ pos_h, const float* __restrict__ pos_w,
    unsigned short* __restrict__ qs, unsigned short* __restrict__ kp,
    unsigned short* __restrict__ vs) {
  __shared__ __align__(16) unsigned short As[2][64 * 32];
  __shared__ __align__(16) unsigned short Bs[2][64 * 32];
  const int tid = threadIdx.x, lane = tid & 31, wave = tid >> 5;
  const int wm = wave & 3, wn = wave >> 2;
  const int r0 = blockIdx.x * 64, c0 = blockIdx.y * 64;

  const unsigned short* at = out1 + (size_t)r0 * MID;   // +32 per iter
  const unsigned short* wt = wcB + (size_t)c0 * 32;     // +QKV*32 per iter
  const unsigned asOff = (unsigned)(size_t)(void*)&As[0][0];
  const unsigned bsOff = (unsigned)(size_t)(void*)&Bs[0][0];

  if (wave == 0) {
    tdm_load(asOff, at, 32, 64, 32, 64, MID);
    tdm_load(bsOff, wt, 2048, 1, 2048, 1, 2048);
  }

  v8f acc0 = v8zero(), acc1 = v8zero();
  for (int k0 = 0; k0 < MID; k0 += 32) {
    const int buf = (k0 >> 5) & 1;
    if (wave == 0) {
      if (k0 + 32 < MID) {
        tdm_load(asOff + (unsigned)(buf ^ 1) * 4096, at + 32, 32, 64, 32, 64, MID);
        tdm_load(bsOff + (unsigned)(buf ^ 1) * 4096, wt + (size_t)QKV * 32,
                 2048, 1, 2048, 1, 2048);
        __builtin_amdgcn_s_wait_tensorcnt(2);
      } else {
        __builtin_amdgcn_s_wait_tensorcnt(0);
      }
    }
    __syncthreads();
    v16bf a = load_a_frag(As[buf], wm * 16, lane);
    acc0 = wmma_bf16(a, load_b_frag(Bs[buf], wn * 32, lane), acc0);
    acc1 = wmma_bf16(a, load_b_frag(Bs[buf], wn * 32 + 16, lane), acc1);
    __syncthreads();
    at += 32;
    wt += (size_t)QKV * 32;
  }
  const int nI = r0 >> 10, hwB = (r0 & 1023) + wm * 16 + 8 * (lane >> 4);
#pragma unroll
  for (int f = 0; f < 2; ++f) {
    const int col = c0 + wn * 32 + f * 16 + (lane & 15);
    v8f acc = f ? acc1 : acc0;
#pragma unroll
    for (int j = 0; j < 8; ++j) {
      const int hw = hwB + j;
      float val = acc[j];
      if (col < 512) {
        int head = col >> 7, d = col & 127;
        qs[((size_t)(nI * 4 + head) * HW + hw) * 128 + d] = f2bf(val * SCALE);
      } else if (col < 1024) {
        int c2 = col - 512, head = c2 >> 7, d = c2 & 127;
        val += pos_h[(hw >> 5) * 128 + d] + pos_w[(hw & 31) * 128 + d];
        kp[((size_t)(nI * 4 + head) * HW + hw) * 128 + d] = f2bf(val);
      } else {
        int c2 = col - 1024, head = c2 >> 7, d = c2 & 127;
        vs[((size_t)(nI * 4 + head) * HW + hw) * 128 + d] = f2bf(val);
      }
    }
  }
}

// ---------------------------------------------------------------------------
// flash attention: per (n,head) softmax(q @ k'^T) @ v, relu -> attn[row][512]
// grid (64, 8); 8 waves x 16 q-rows; K'/V tiles double-buffered via TDM.
// ---------------------------------------------------------------------------
__global__ __launch_bounds__(256) void attn_kernel(
    const unsigned short* __restrict__ qs, const unsigned short* __restrict__ kp,
    const unsigned short* __restrict__ vs, unsigned short* __restrict__ attn) {
  __shared__ __align__(16) unsigned short Ks[2][32 * 128];   // [y][d]
  __shared__ __align__(16) unsigned short Vl[2][32 * 128];   // [y][d]
  __shared__ __align__(16) unsigned short Ps[8 * 16 * 32];
  const int tid = threadIdx.x, lane = tid & 31, wave = tid >> 5;
  const int nh = blockIdx.x;
  const int rw = blockIdx.y * 128 + wave * 16;
  const size_t base = (size_t)nh * HW * 128;
  const unsigned ksOff = (unsigned)(size_t)(void*)&Ks[0][0];
  const unsigned vlOff = (unsigned)(size_t)(void*)&Vl[0][0];

  if (wave == 0) {
    tdm_load(ksOff, kp + base, 4096, 1, 4096, 1, 4096);
    tdm_load(vlOff, vs + base, 4096, 1, 4096, 1, 4096);
  }

  // Q fragments (whole 128-d row) in VGPRs
  AFrag qf[4];
  {
    const unsigned short* qrow = qs + base + (size_t)(rw + (lane & 15)) * 128;
    const int kb = (lane >> 4) * 8;
#pragma unroll
    for (int ks = 0; ks < 4; ++ks)
#pragma unroll
      for (int v = 0; v < 8; ++v) {
        int k = ks * 32 + ((v < 4) ? (2 * v) : (16 + 2 * (v - 4))) + kb;
        qf[ks].d[v] = *(const unsigned int*)(qrow + k);
      }
  }

  v8f acc[8];
#pragma unroll
  for (int i = 0; i < 8; ++i) acc[i] = v8zero();
  float run_max[8], run_sum[8];
#pragma unroll
  for (int j = 0; j < 8; ++j) { run_max[j] = -1e30f; run_sum[j] = 0.f; }

  for (int y0 = 0; y0 < HW; y0 += 32) {
    const int buf = (y0 >> 5) & 1;
    if (wave == 0) {
      if (y0 + 32 < HW) {
        tdm_load(ksOff + (unsigned)(buf ^ 1) * 8192,
                 kp + base + (size_t)(y0 + 32) * 128, 4096, 1, 4096, 1, 4096);
        tdm_load(vlOff + (unsigned)(buf ^ 1) * 8192,
                 vs + base + (size_t)(y0 + 32) * 128, 4096, 1, 4096, 1, 4096);
        __builtin_amdgcn_s_wait_tensorcnt(2);
      } else {
        __builtin_amdgcn_s_wait_tensorcnt(0);
      }
    }
    __syncthreads();

    // S(16x32) = Q @ K'^T
    v8f s0 = v8zero(), s1 = v8zero();
#pragma unroll
    for (int ks = 0; ks < 4; ++ks) {
      AFrag b0, b1;
      const int n0 = lane & 15;
      const int kd = (lane >> 4) * 4 + ks * 16;
      const unsigned int* r0p = (const unsigned int*)(Ks[buf] + (size_t)n0 * 128);
      const unsigned int* r1p = (const unsigned int*)(Ks[buf] + (size_t)(n0 + 16) * 128);
#pragma unroll
      for (int v = 0; v < 8; ++v) { const int q = kd + v + ((v < 4) ? 0 : 4); b0.d[v] = r0p[q]; b1.d[v] = r1p[q]; }
      s0 = wmma_bf16(qf[ks].v, b0.v, s0);
      s1 = wmma_bf16(qf[ks].v, b1.v, s1);
    }

    // online softmax: row m = j + 8*(lane>>4), reduce across 16-lane half
#pragma unroll
    for (int j = 0; j < 8; ++j) {
      float mx = fmaxf(s0[j], s1[j]);
#pragma unroll
      for (int off = 8; off >= 1; off >>= 1) mx = fmaxf(mx, __shfl_xor(mx, off, 32));
      const float nm = fmaxf(run_max[j], mx);
      const float corr = __expf(run_max[j] - nm);
      run_max[j] = nm;
      const float p0 = __expf(s0[j] - nm), p1 = __expf(s1[j] - nm);
      float rs = p0 + p1;
#pragma unroll
      for (int off = 8; off >= 1; off >>= 1) rs += __shfl_xor(rs, off, 32);
      run_sum[j] = run_sum[j] * corr + rs;
      s0[j] = p0; s1[j] = p1;
#pragma unroll
      for (int db = 0; db < 8; ++db) acc[db][j] *= corr;
    }

    // D-layout -> A-layout for P via per-wave LDS round trip
    unsigned short* pw = Ps + (size_t)wave * 16 * 32;
    {
      const int n0 = lane & 15, hi = lane >> 4;
#pragma unroll
      for (int j = 0; j < 8; ++j) {
        const int m = j + 8 * hi;
        pw[m * 32 + n0]      = f2bf(s0[j]);
        pw[m * 32 + 16 + n0] = f2bf(s1[j]);
      }
    }
    AFrag pf;
    {
      const unsigned int* row = (const unsigned int*)(pw + (size_t)(lane & 15) * 32);
      const int kb = (lane >> 4) * 16;
#pragma unroll
      for (int v = 0; v < 8; ++v) {
        int k = 2 * v + kb;
        pf.d[v] = row[k >> 1];
      }
    }

    // acc += P(16x32) @ V(32x128); B gathered strided from natural [y][d]
#pragma unroll
    for (int db = 0; db < 8; ++db) {
      AFrag bf;
      const int dcol = db * 16 + (lane & 15), kb = (lane >> 4) * 16;
#pragma unroll
      for (int v = 0; v < 8; ++v) {
        unsigned lo = Vl[buf][(size_t)(kb + 2 * v) * 128 + dcol];
        unsigned hi = Vl[buf][(size_t)(kb + 2 * v + 1) * 128 + dcol];
        bf.d[v] = lo | (hi << 16);
      }
      acc[db] = wmma_bf16(pf.v, bf.v, acc[db]);
    }
    __syncthreads();
  }

  // epilogue: normalize, relu, store bf16 to attn[row][512]
  const int nI = nh >> 2, head = nh & 3;
  const int n0 = lane & 15, hi = lane >> 4;
#pragma unroll
  for (int j = 0; j < 8; ++j) {
    const int row = nI * HW + rw + j + 8 * hi;
    const float inv = 1.0f / run_sum[j];
#pragma unroll
    for (int db = 0; db < 8; ++db) {
      float val = acc[db][j] * inv;
      attn[(size_t)row * MID + head * 128 + db * 16 + n0] = f2bf(val > 0.f ? val : 0.f);
    }
  }
}

// ---------------------------------------------------------------------------
// conv3: out = relu(bn3(attn @ w3^T) + x), f32 NCHW.  grid (256, 32)
// ---------------------------------------------------------------------------
__global__ __launch_bounds__(256) void conv3_gemm(
    const unsigned short* __restrict__ attn, const unsigned short* __restrict__ w3B,
    const float* __restrict__ g, const float* __restrict__ bta,
    const float* __restrict__ mn, const float* __restrict__ vr,
    const float* __restrict__ x, float* __restrict__ out) {
  __shared__ __align__(16) unsigned short As[2][64 * 32];
  __shared__ __align__(16) unsigned short Bs[2][64 * 32];
  const int tid = threadIdx.x, lane = tid & 31, wave = tid >> 5;
  const int wm = wave & 3, wn = wave >> 2;
  const int r0 = blockIdx.x * 64, c0 = blockIdx.y * 64;

  const unsigned short* at = attn + (size_t)r0 * MID;
  const unsigned short* wt = w3B + (size_t)c0 * 32;
  const unsigned asOff = (unsigned)(size_t)(void*)&As[0][0];
  const unsigned bsOff = (unsigned)(size_t)(void*)&Bs[0][0];

  if (wave == 0) {
    tdm_load(asOff, at, 32, 64, 32, 64, MID);
    tdm_load(bsOff, wt, 2048, 1, 2048, 1, 2048);
  }

  v8f acc0 = v8zero(), acc1 = v8zero();
  for (int k0 = 0; k0 < MID; k0 += 32) {
    const int buf = (k0 >> 5) & 1;
    if (wave == 0) {
      if (k0 + 32 < MID) {
        tdm_load(asOff + (unsigned)(buf ^ 1) * 4096, at + 32, 32, 64, 32, 64, MID);
        tdm_load(bsOff + (unsigned)(buf ^ 1) * 4096, wt + (size_t)CIN * 32,
                 2048, 1, 2048, 1, 2048);
        __builtin_amdgcn_s_wait_tensorcnt(2);
      } else {
        __builtin_amdgcn_s_wait_tensorcnt(0);
      }
    }
    __syncthreads();
    v16bf a = load_a_frag(As[buf], wm * 16, lane);
    acc0 = wmma_bf16(a, load_b_frag(Bs[buf], wn * 32, lane), acc0);
    acc1 = wmma_bf16(a, load_b_frag(Bs[buf], wn * 32 + 16, lane), acc1);
    __syncthreads();
    at += 32;
    wt += (size_t)CIN * 32;
  }
  const int nI = r0 >> 10;
  const int hwB = (r0 & 1023) + wm * 16 + 8 * (lane >> 4);
#pragma unroll
  for (int f = 0; f < 2; ++f) {
    const int col = c0 + wn * 32 + f * 16 + (lane & 15);
    const float inv = g[col] * rsqrtf(vr[col] + 1e-5f);
    const float add = bta[col] - mn[col] * inv;
    v8f acc = f ? acc1 : acc0;
#pragma unroll
    for (int j = 0; j < 8; ++j) {
      const size_t oidx = (size_t)nI * CIN * HW + (size_t)col * HW + (hwB + j);
      float val = acc[j] * inv + add + x[oidx];
      out[oidx] = val > 0.f ? val : 0.f;
    }
  }
}

// ---------------------------------------------------------------------------
// launcher
// ---------------------------------------------------------------------------
extern "C" void kernel_launch(void* const* d_in, const int* in_sizes, int n_in,
                              void* d_out, int out_size, void* d_ws, size_t ws_size,
                              hipStream_t stream) {
  const float* x       = (const float*)d_in[0];
  const float* conv1_w = (const float*)d_in[1];
  const float* gamma1  = (const float*)d_in[2];
  const float* beta1   = (const float*)d_in[3];
  const float* mean1   = (const float*)d_in[4];
  const float* var1    = (const float*)d_in[5];
  const float* qk_w    = (const float*)d_in[6];
  const float* v_w     = (const float*)d_in[7];
  const float* pos_h   = (const float*)d_in[8];
  const float* pos_w   = (const float*)d_in[9];
  const float* conv3_w = (const float*)d_in[10];
  const float* gamma3  = (const float*)d_in[11];
  const float* beta3   = (const float*)d_in[12];
  const float* mean3   = (const float*)d_in[13];
  const float* var3    = (const float*)d_in[14];
  float* out = (float*)d_out;

  char* ws = (char*)d_ws;
  unsigned short* W1B  = (unsigned short*)(ws + 0);            // 2048x512 blocked
  unsigned short* WCB  = (unsigned short*)(ws + 2097152u);     // 512x1536 blocked
  unsigned short* W3B  = (unsigned short*)(ws + 3670016u);     // 512x2048 blocked
  unsigned short* OUT1 = (unsigned short*)(ws + 8388608u);     // 16384x512 (reused as attn)
  unsigned short* QS   = (unsigned short*)(ws + 25165824u);    // 64x1024x128
  unsigned short* KP   = (unsigned short*)(ws + 41943040u);
  unsigned short* VS   = (unsigned short*)(ws + 58720256u);

  wcvt_kernel<<<dim3((512 * 2048 + 255) / 256), dim3(256), 0, stream>>>(
      conv1_w, W1B, 512, 2048, 512, 0);
  wcvt_kernel<<<dim3((1024 * 512 + 255) / 256), dim3(256), 0, stream>>>(
      qk_w, WCB, 1024, 512, 1536, 0);
  wcvt_kernel<<<dim3((512 * 512 + 255) / 256), dim3(256), 0, stream>>>(
      v_w, WCB, 512, 512, 1536, 1024);
  wcvt_kernel<<<dim3((2048 * 512 + 255) / 256), dim3(256), 0, stream>>>(
      conv3_w, W3B, 2048, 512, 2048, 0);

  conv1_gemm<<<dim3(NROWS / 64, MID / 64), dim3(256), 0, stream>>>(
      x, W1B, gamma1, beta1, mean1, var1, OUT1);

  qkv_gemm<<<dim3(NROWS / 64, QKV / 64), dim3(256), 0, stream>>>(
      OUT1, WCB, pos_h, pos_w, QS, KP, VS);

  attn_kernel<<<dim3(NIMG * 4, HW / 128), dim3(256), 0, stream>>>(
      QS, KP, VS, OUT1);

  conv3_gemm<<<dim3(NROWS / 64, CIN / 64), dim3(256), 0, stream>>>(
      OUT1, W3B, gamma3, beta3, mean3, var3, x, out);
}
